// PraxisMRU_29283087024359
// MI455X (gfx1250) — matrix operand order repaired and checked by reading, not compiled, not run on hardware
//
#include <hip/hip_runtime.h>
#include <cstdint>
#include <cstddef>

// ---------------- types ----------------
typedef __attribute__((ext_vector_type(16))) __bf16 v16bf;
typedef __attribute__((ext_vector_type(8)))  __bf16 v8bf;
typedef __attribute__((ext_vector_type(8)))  float  v8f;

#define B_    8
#define S_    2048
#define E_    1024
#define H_    16
#define DFF_  4096
#define NTOK  (B_*S_)          // 16384
#define NC_   16               // scan chunks per (b,h)
#define CL_   128              // chunk length (NC_*CL_ == S_)
#define SCALE_UP_ 0.16f
#define OUT_SCALE_ 128.0f

__device__ __forceinline__ unsigned short f2bf(float f) {
  union { float f; unsigned u; } v; v.f = f;
  unsigned r = v.u + 0x7FFFu + ((v.u >> 16) & 1u);   // round-nearest-even
  return (unsigned short)(r >> 16);
}

union AFrag { v16bf v; v8bf h[2]; };

__device__ __forceinline__ v8f wmma_bf16(v16bf a, v16bf b, v8f c) {
  return __builtin_amdgcn_wmma_f32_16x16x32_bf16(false, a, false, b, (short)0, c, false, false);
}

// A fragment (16x32 bf16): lane half h pulls 16 contiguous k (k = 16h + s)
__device__ __forceinline__ v16bf load_afrag(const unsigned short* arow, int k) {
  AFrag a;
  a.h[0] = *reinterpret_cast<const v8bf*>(arow + k);
  a.h[1] = *reinterpret_cast<const v8bf*>(arow + k + 8);
  return a.v;
}
// B fragment (32x16 bf16): 32 contiguous bytes of the weight row
__device__ __forceinline__ v16bf load_bfrag(const unsigned short* p) {
  return *reinterpret_cast<const v16bf*>(p);
}

// ---------------- fp32 -> bf16 convert ----------------
__global__ void cvt_bf16_kernel(const float* __restrict__ src,
                                unsigned short* __restrict__ dst, int n) {
  int i = blockIdx.x * blockDim.x + threadIdx.x;
  if (i < n) dst[i] = f2bf(src[i]);
}

// ---------------- LN1 + per-head transition matrices ----------------
__global__ __launch_bounds__(256) void ln1_makeM_kernel(
    const float* __restrict__ x, const float* __restrict__ g,
    const float* __restrict__ up, float* __restrict__ Mbuf) {
  __shared__ float2 red[256];
  __shared__ float  hn[E_];
  const int tid = threadIdx.x;
  const size_t token = blockIdx.x;
  const float4 v4 = reinterpret_cast<const float4*>(x + token * E_)[tid];
  float s = v4.x + v4.y + v4.z + v4.w;
  float q = v4.x*v4.x + v4.y*v4.y + v4.z*v4.z + v4.w*v4.w;
  red[tid] = make_float2(s, q);
  __syncthreads();
  for (int off = 128; off > 0; off >>= 1) {
    if (tid < off) { red[tid].x += red[tid+off].x; red[tid].y += red[tid+off].y; }
    __syncthreads();
  }
  const float mu  = red[0].x * (1.0f / E_);
  const float var = red[0].y * (1.0f / E_) - mu * mu;
  const float inv = rsqrtf(var + 1e-5f);
  hn[4*tid+0] = (v4.x - mu) * inv * g[4*tid+0];
  hn[4*tid+1] = (v4.y - mu) * inv * g[4*tid+1];
  hn[4*tid+2] = (v4.z - mu) * inv * g[4*tid+2];
  hn[4*tid+3] = (v4.w - mu) * inv * g[4*tid+3];
  __syncthreads();
  for (int idx = tid; idx < E_; idx += 256) {
    const int o = (idx >> 3) & 7, p = idx & 7;
    float acc = 0.0f;
    #pragma unroll
    for (int c = 0; c < 8; ++c)
      acc += hn[(idx & ~7) + c] * up[(idx & 0x3C0) + c * 8 + (idx & 7)];
    Mbuf[token * E_ + idx] = acc * SCALE_UP_ + ((o == p) ? 1.0f : 0.0f);
  }
}

// ---------------- LN2 -> bf16 ----------------
__global__ __launch_bounds__(256) void ln2_bf16_kernel(
    const float* __restrict__ x, const float* __restrict__ g,
    unsigned short* __restrict__ h2) {
  __shared__ float2 red[256];
  const int tid = threadIdx.x;
  const size_t token = blockIdx.x;
  const float4 v4 = reinterpret_cast<const float4*>(x + token * E_)[tid];
  float s = v4.x + v4.y + v4.z + v4.w;
  float q = v4.x*v4.x + v4.y*v4.y + v4.z*v4.z + v4.w*v4.w;
  red[tid] = make_float2(s, q);
  __syncthreads();
  for (int off = 128; off > 0; off >>= 1) {
    if (tid < off) { red[tid].x += red[tid+off].x; red[tid].y += red[tid+off].y; }
    __syncthreads();
  }
  const float mu  = red[0].x * (1.0f / E_);
  const float var = red[0].y * (1.0f / E_) - mu * mu;
  const float inv = rsqrtf(var + 1e-5f);
  unsigned short* o = h2 + token * E_ + 4 * tid;
  o[0] = f2bf((v4.x - mu) * inv * g[4*tid+0]);
  o[1] = f2bf((v4.y - mu) * inv * g[4*tid+1]);
  o[2] = f2bf((v4.z - mu) * inv * g[4*tid+2]);
  o[3] = f2bf((v4.w - mu) * inv * g[4*tid+3]);
}

// ---------------- chunked matrix-product scan ----------------
__global__ __launch_bounds__(64) void scan_phase1_kernel(
    const float* __restrict__ Mbuf, float* __restrict__ ctot) {
  __shared__ float P[64], Mt[64];
  const int tid = threadIdx.x;
  const int bid = blockIdx.x;
  const int chunk = bid & (NC_ - 1);
  const int bh = bid >> 4;            // NC_ == 16
  const int b = bh >> 4, h = bh & 15; // H_ == 16
  const int o = tid >> 3, p = tid & 7;
  const size_t base   = (((size_t)b * S_ + (size_t)chunk * CL_) * H_ + h) * 64;
  const size_t stride = (size_t)H_ * 64;
  P[tid] = (o == p) ? 1.0f : 0.0f;
  for (int t = 0; t < CL_; ++t) {
    Mt[tid] = Mbuf[base + (size_t)t * stride + tid];
    __syncthreads();
    float nv = 0.0f;
    #pragma unroll
    for (int c = 0; c < 8; ++c) nv += P[o*8 + c] * Mt[c*8 + p];
    __syncthreads();
    P[tid] = nv;
  }
  __syncthreads();
  ctot[(size_t)bid * 64 + tid] = P[tid];
}

__global__ __launch_bounds__(64) void scan_phase2_kernel(
    const float* __restrict__ ctot, float* __restrict__ epre) {
  __shared__ float Ew[64], Ct[64];
  const int tid = threadIdx.x;
  const int bh = blockIdx.x;
  const int o = tid >> 3, p = tid & 7;
  Ew[tid] = (o == p) ? 1.0f : 0.0f;
  for (int i = 0; i < NC_; ++i) {
    const size_t idx = ((size_t)bh * NC_ + i) * 64 + tid;
    epre[idx] = Ew[tid];
    Ct[tid] = ctot[idx];
    __syncthreads();
    float nv = 0.0f;
    #pragma unroll
    for (int c = 0; c < 8; ++c) nv += Ew[o*8 + c] * Ct[c*8 + p];
    __syncthreads();
    Ew[tid] = nv;
  }
}

__global__ __launch_bounds__(64) void scan_phase3_kernel(
    float* __restrict__ Mbuf, const float* __restrict__ epre) {
  __shared__ float P[64], Mt[64];
  const int tid = threadIdx.x;
  const int bid = blockIdx.x;
  const int chunk = bid & (NC_ - 1);
  const int bh = bid >> 4;
  const int b = bh >> 4, h = bh & 15;
  const int o = tid >> 3, p = tid & 7;
  const size_t base   = (((size_t)b * S_ + (size_t)chunk * CL_) * H_ + h) * 64;
  const size_t stride = (size_t)H_ * 64;
  P[tid] = epre[(size_t)bid * 64 + tid];
  for (int t = 0; t < CL_; ++t) {
    Mt[tid] = Mbuf[base + (size_t)t * stride + tid];
    __syncthreads();
    float nv = 0.0f;
    #pragma unroll
    for (int c = 0; c < 8; ++c) nv += P[o*8 + c] * Mt[c*8 + p];
    __syncthreads();
    P[tid] = nv;
    Mbuf[base + (size_t)t * stride + tid] = nv;   // states in place over M
  }
}

// ---------------- y = OUT_SCALE * states @ down  (bf16) ----------------
__global__ __launch_bounds__(256) void compute_y_kernel(
    const float* __restrict__ states, const float* __restrict__ down,
    unsigned short* __restrict__ ybf) {
  const size_t idx = (size_t)blockIdx.x * 256 + threadIdx.x;   // over B*S*E
  const int e = (int)(idx & (E_ - 1));
  const size_t token = idx >> 10;
  const int h = e >> 6, c = e & 7;
  const float* st = states + token * E_ + (e & ~7);
  const float* dn = down + h * 64 + c;        // down[h][p][c]
  float s = 0.0f;
  #pragma unroll
  for (int p = 0; p < 8; ++p) s += st[p] * dn[p * 8];
  ybf[idx] = f2bf(s * OUT_SCALE_);
}

__global__ void copy_states_kernel(const float* __restrict__ states,
                                   float* __restrict__ out_st) {
  const size_t idx = (size_t)blockIdx.x * 256 + threadIdx.x;   // over S_*H_*64
  out_st[idx] = states[(size_t)(B_ - 1) * S_ * H_ * 64 + idx];
}

// ---------------- WMMA GEMM: C = A(bf16)[MxK] @ W(bf16)[N rows, ldw]^T ----------------
// Wave tile 32x64 (2 A-frags x 4 B-frags = 8 WMMA / k-step), software-pipelined
// double-buffered fragments so loads for k+32 overlap WMMAs at k.
// MODE 0: Out(f32) = Res + acc ; MODE 1: Out(bf16) = gelu(acc) ; MODE 2: Out(f32) += acc
template<int MODE>
__global__ __launch_bounds__(256) void wmma_gemm_kernel(
    const unsigned short* __restrict__ A, const unsigned short* __restrict__ W,
    const float* __restrict__ Res, void* __restrict__ Out,
    int M, int N, int K, int ldw) {
  const int lane   = threadIdx.x & 31;
  const int waveId = threadIdx.x >> 5;
  const int mbase  = blockIdx.y * 64  + (waveId & 1) * 32;   // 2 waves in M
  const int nbase  = blockIdx.x * 256 + (waveId >> 1) * 64;  // 4 waves in N
  const int laneHi = lane >> 4;
  const int lrow   = lane & 15;

  v8f acc[2][4] = {};

  const unsigned short* arow0 = A + (size_t)(mbase + lrow) * K      + (laneHi ? 16 : 0);
  const unsigned short* arow1 = A + (size_t)(mbase + 16 + lrow) * K + (laneHi ? 16 : 0);
  const unsigned short* wrow[4];
  #pragma unroll
  for (int j = 0; j < 4; ++j)
    wrow[j] = W + (size_t)(nbase + j * 16 + lrow) * (size_t)ldw + (size_t)laneHi * 16;

  // ---- software pipeline: ping (pA) / pong (pB) fragment buffers ----
  v16bf a0A = load_afrag(arow0, 0);
  v16bf a1A = load_afrag(arow1, 0);
  v16bf bA[4], bB[4];
  #pragma unroll
  for (int j = 0; j < 4; ++j) bA[j] = load_bfrag(wrow[j]);
  v16bf a0B, a1B;

  int k = 0;
  for (; k < K - 64; k += 64) {
    // prefetch k+32 into pong
    a0B = load_afrag(arow0, k + 32);
    a1B = load_afrag(arow1, k + 32);
    #pragma unroll
    for (int j = 0; j < 4; ++j) bB[j] = load_bfrag(wrow[j] + k + 32);
    // consume ping (k)
    #pragma unroll
    for (int j = 0; j < 4; ++j) {
      acc[0][j] = wmma_bf16(a0A, bA[j], acc[0][j]);
      acc[1][j] = wmma_bf16(a1A, bA[j], acc[1][j]);
    }
    // prefetch k+64 into ping
    a0A = load_afrag(arow0, k + 64);
    a1A = load_afrag(arow1, k + 64);
    #pragma unroll
    for (int j = 0; j < 4; ++j) bA[j] = load_bfrag(wrow[j] + k + 64);
    // consume pong (k+32)
    #pragma unroll
    for (int j = 0; j < 4; ++j) {
      acc[0][j] = wmma_bf16(a0B, bB[j], acc[0][j]);
      acc[1][j] = wmma_bf16(a1B, bB[j], acc[1][j]);
    }
  }
  // tail: k == K-64; ping holds k
  a0B = load_afrag(arow0, k + 32);
  a1B = load_afrag(arow1, k + 32);
  #pragma unroll
  for (int j = 0; j < 4; ++j) bB[j] = load_bfrag(wrow[j] + k + 32);
  #pragma unroll
  for (int j = 0; j < 4; ++j) {
    acc[0][j] = wmma_bf16(a0A, bA[j], acc[0][j]);
    acc[1][j] = wmma_bf16(a1A, bA[j], acc[1][j]);
  }
  #pragma unroll
  for (int j = 0; j < 4; ++j) {
    acc[0][j] = wmma_bf16(a0B, bB[j], acc[0][j]);
    acc[1][j] = wmma_bf16(a1B, bB[j], acc[1][j]);
  }

  // ---- store ----
  float* outf = reinterpret_cast<float*>(Out);
  unsigned short* outb = reinterpret_cast<unsigned short*>(Out);
  #pragma unroll
  for (int i = 0; i < 2; ++i) {
    const int om = mbase + i * 16 + laneHi * 8;
    #pragma unroll
    for (int j = 0; j < 4; ++j) {
      const int on = nbase + j * 16 + lrow;
      #pragma unroll
      for (int r = 0; r < 8; ++r) {
        const size_t oidx = (size_t)(om + r) * N + on;
        const float v = acc[i][j][r];
        if (MODE == 0) {
          outf[oidx] = Res[oidx] + v;
        } else if (MODE == 1) {
          const float gelu = 0.5f * v * (1.0f + erff(v * 0.70710678118654752f));
          outb[oidx] = f2bf(gelu);
        } else {
          outf[oidx] += v;
        }
      }
    }
  }
}

// ---------------- host launch ----------------
extern "C" void kernel_launch(void* const* d_in, const int* in_sizes, int n_in,
                              void* d_out, int out_size, void* d_ws, size_t ws_size,
                              hipStream_t stream) {
  (void)in_sizes; (void)n_in; (void)out_size; (void)ws_size;
  const float* x        = (const float*)d_in[0];
  const float* up       = (const float*)d_in[1];
  const float* down     = (const float*)d_in[2];
  const float* w_out    = (const float*)d_in[3];
  const float* mru_w    = (const float*)d_in[4];
  const float* ffn_w    = (const float*)d_in[5];
  const float* ffn_up   = (const float*)d_in[6];
  const float* ffn_down = (const float*)d_in[7];

  char* ws = (char*)d_ws;
  float*          Mbuf    = (float*)(ws + 0);                  // 67,108,864  (M -> states; later mid bf16 halves)
  float*          ctot    = (float*)(ws + 67108864);           //    524,288
  float*          epre    = (float*)(ws + 67633152);           //    524,288
  unsigned short* ybf     = (unsigned short*)(ws + 68157440);  // 33,554,432  (y bf16; later h2 bf16)
  unsigned short* h2      = ybf;
  unsigned short* wout_bf = (unsigned short*)(ws + 101711872); //  2,097,152
  unsigned short* fup_bf  = (unsigned short*)(ws + 103809024); //  8,388,608
  unsigned short* fdn_bf  = (unsigned short*)(ws + 112197632); //  8,388,608
  unsigned short* midbf   = (unsigned short*)Mbuf;             // 16384 x 2048 bf16

  float* out_x  = (float*)d_out;
  float* out_st = out_x + (size_t)B_ * S_ * E_;

  // 1) weights -> bf16
  cvt_bf16_kernel<<<(E_*E_ + 255)/256, 256, 0, stream>>>(w_out, wout_bf, E_*E_);
  cvt_bf16_kernel<<<(DFF_*E_ + 255)/256, 256, 0, stream>>>(ffn_up, fup_bf, DFF_*E_);
  cvt_bf16_kernel<<<(E_*DFF_ + 255)/256, 256, 0, stream>>>(ffn_down, fdn_bf, E_*DFF_);

  // 2) LN1 + transition matrices
  ln1_makeM_kernel<<<NTOK, 256, 0, stream>>>(x, mru_w, up, Mbuf);

  // 3) chunked matrix-product scan (states written over Mbuf)
  scan_phase1_kernel<<<B_*H_*NC_, 64, 0, stream>>>(Mbuf, ctot);
  scan_phase2_kernel<<<B_*H_, 64, 0, stream>>>(ctot, epre);
  scan_phase3_kernel<<<B_*H_*NC_, 64, 0, stream>>>(Mbuf, epre);

  // 4) states[-1] to output tail; y = states @ down
  copy_states_kernel<<<(S_*H_*64)/256, 256, 0, stream>>>(Mbuf, out_st);
  compute_y_kernel<<<(NTOK*E_)/256, 256, 0, stream>>>(Mbuf, down, ybf);

  // 5) GEMM1: out_x = x + y @ w_out^T   [16384 x 1024 x 1024]
  wmma_gemm_kernel<0><<<dim3(E_/256, NTOK/64), 256, 0, stream>>>(
      ybf, wout_bf, x, out_x, NTOK, E_, E_, E_);

  // 6) LN2 -> h2 (bf16)
  ln2_bf16_kernel<<<NTOK, 256, 0, stream>>>(out_x, ffn_w, h2);

  // 7) FFN in two DFF halves (mid reuses Mbuf region)
  wmma_gemm_kernel<1><<<dim3(2048/256, NTOK/64), 256, 0, stream>>>(
      h2, fup_bf, nullptr, midbf, NTOK, 2048, E_, E_);
  wmma_gemm_kernel<2><<<dim3(E_/256, NTOK/64), 256, 0, stream>>>(
      midbf, fdn_bf, nullptr, out_x, NTOK, E_, 2048, DFF_);
  wmma_gemm_kernel<1><<<dim3(2048/256, NTOK/64), 256, 0, stream>>>(
      h2, fup_bf + (size_t)2048 * E_, nullptr, midbf, NTOK, 2048, E_, E_);
  wmma_gemm_kernel<2><<<dim3(E_/256, NTOK/64), 256, 0, stream>>>(
      midbf, fdn_bf + 2048, nullptr, out_x, NTOK, E_, 2048, DFF_);
}
